// MicroGPTTorch_69647189672022
// MI455X (gfx1250) — matrix operand order repaired and checked by reading, not compiled, not run on hardware
//
#include <hip/hip_runtime.h>
#include <hip/hip_bf16.h>
#include <math.h>

#define D_MODEL 768
#define N_HEADS 12
#define D_HEAD  64
#define D_FF    3072
#define N_LAYERS 4
#define VOCAB   32000
#define BATCH   4
#define SEQ     1024
#define NEGV    (-1.0e9f)

typedef __bf16 bf16_t;
typedef __attribute__((ext_vector_type(16))) __bf16 v16bf;
typedef __attribute__((ext_vector_type(8)))  float  v8f;
typedef __attribute__((ext_vector_type(4)))  unsigned int u32x4v;
typedef __attribute__((ext_vector_type(8)))  int  i32x8v;
typedef __attribute__((ext_vector_type(4)))  int  i32x4v;

union Frag {
    v16bf  v;
    uint4  q[2];
    bf16_t h[16];
};

__device__ __forceinline__ v8f wmma_bf16(const Frag& a, const Frag& b, v8f c) {
    // 8 args: (neg_a, A, neg_b, B, c_mod, C, reuse_a, reuse_b)
    return __builtin_amdgcn_wmma_f32_16x16x32_bf16(false, a.v, false, b.v,
                                                   (short)0, c, false, false);
}

// ---------------------------------------------------------------------------
// Tensor Data Mover: DMA a 2-D bf16 tile (tile_w x tile_h) whose rows are
// strided by row_stride elements, from global memory into LDS at byte offset
// lds_byte_off.  D# per CDNA5 ISA ch.8: group0 = {flags, lds_addr,
// global_addr, type=2}, group1 = {data_size=2B, tensor dims, tile dims,
// stride}; groups 2/3 zero (2-D tile, tile_dim2 = 0).
// ---------------------------------------------------------------------------
__device__ __forceinline__ void tdm_load_tile_2d(unsigned int lds_byte_off,
                                                 const bf16_t* gsrc,
                                                 unsigned int row_len,     // tensor_dim0 (elems)
                                                 unsigned int row_stride,  // dim0 stride (elems)
                                                 unsigned int nrows,       // tensor_dim1
                                                 unsigned int tile_w,
                                                 unsigned int tile_h) {
    unsigned long long ga = (unsigned long long)(uintptr_t)gsrc;
    u32x4v g0;
    g0.x = 1u;                                            // count=1, user mode
    g0.y = lds_byte_off;                                  // lds_addr
    g0.z = (unsigned int)ga;                              // global_addr[31:0]
    g0.w = ((unsigned int)(ga >> 32) & 0x01FFFFFFu)       // global_addr[56:32]
           | (2u << 30);                                  // type=2 ("image")
    i32x8v g1;
    g1[0] = (int)(1u << 16);                              // wg_mask=0, data_size=1 (2B)
    g1[1] = (int)((row_len & 0xFFFFu) << 16);             // tensor_dim0[15:0] @ bits 63:48
    g1[2] = (int)(((row_len >> 16) & 0xFFFFu)             // tensor_dim0[31:16]
           | ((nrows & 0xFFFFu) << 16));                  // tensor_dim1[15:0]
    g1[3] = (int)(((nrows >> 16) & 0xFFFFu)               // tensor_dim1[31:16]
           | ((tile_w & 0xFFFFu) << 16));                 // tile_dim0
    g1[4] = (int)(tile_h & 0xFFFFu);                      // tile_dim1; tile_dim2=0
    g1[5] = (int)row_stride;                              // tensor_dim0_stride[31:0]
    g1[6] = 0;
    g1[7] = 0;
    i32x4v gz = {0, 0, 0, 0};
#if defined(__clang_major__) && __clang_major__ >= 23
    i32x8v gz8 = {0, 0, 0, 0, 0, 0, 0, 0};
    __builtin_amdgcn_tensor_load_to_lds(g0, g1, gz, gz, gz8, 0);
#else
    __builtin_amdgcn_tensor_load_to_lds(g0, g1, gz, gz, 0);
#endif
}

// ---------------------------------------------------------------------------
// float -> bf16 conversion
// ---------------------------------------------------------------------------
__global__ void k_to_bf16(const float* __restrict__ src, bf16_t* __restrict__ dst, int n) {
    int i = blockIdx.x * blockDim.x + threadIdx.x;
    if (i < n) dst[i] = (bf16_t)src[i];
}

// ---------------------------------------------------------------------------
// token embedding gather + sinusoidal positional encoding
// ---------------------------------------------------------------------------
__global__ void k_embed(const int* __restrict__ tokens, const float* __restrict__ emb,
                        float* __restrict__ x) {
    int idx = blockIdx.x * blockDim.x + threadIdx.x;
    if (idx >= BATCH * SEQ * D_MODEL) return;
    int d  = idx % D_MODEL;
    int bs = idx / D_MODEL;
    int s  = bs % SEQ;
    int tok = tokens[bs];
    float e = emb[(size_t)d * VOCAB + tok];
    float expo  = (float)(2 * (d / 2)) / (float)D_MODEL;
    float angle = (float)s / __powf(10000.0f, expo);
    float pe    = (d & 1) ? __cosf(angle) : __sinf(angle);
    x[idx] = e + pe;
}

// ---------------------------------------------------------------------------
// LayerNorm: fp32 row (768) -> bf16 row.  One 256-thread block per row.
// ---------------------------------------------------------------------------
__global__ void k_layernorm(const float* __restrict__ x, const float* __restrict__ g,
                            const float* __restrict__ be, bf16_t* __restrict__ out) {
    __shared__ float red[256];
    const int row = blockIdx.x;
    const int tid = threadIdx.x;
    const float* xr = x + (size_t)row * D_MODEL;

    float v[3];
    float s = 0.f;
    #pragma unroll
    for (int i = 0; i < 3; ++i) { v[i] = xr[tid + i * 256]; s += v[i]; }
    red[tid] = s;
    __syncthreads();
    for (int st = 128; st > 0; st >>= 1) {
        if (tid < st) red[tid] += red[tid + st];
        __syncthreads();
    }
    float mean = red[0] * (1.0f / D_MODEL);
    __syncthreads();

    float s2 = 0.f;
    #pragma unroll
    for (int i = 0; i < 3; ++i) { float dd = v[i] - mean; s2 += dd * dd; }
    red[tid] = s2;
    __syncthreads();
    for (int st = 128; st > 0; st >>= 1) {
        if (tid < st) red[tid] += red[tid + st];
        __syncthreads();
    }
    float rstd = rsqrtf(red[0] * (1.0f / D_MODEL) + 1e-5f);

    bf16_t* orow = out + (size_t)row * D_MODEL;
    #pragma unroll
    for (int i = 0; i < 3; ++i) {
        int c = tid + i * 256;
        orow[c] = (bf16_t)(g[c] * ((v[i] - mean) * rstd) + be[c]);
    }
}

// ---------------------------------------------------------------------------
// WMMA GEMM with TDM double-buffered LDS staging, 64-deep k-stages.
//   C[M,N] = A[M,K] * W[N,K]^T   (bf16 in, f32 accum)
// mode 0: Cb = result; mode 1: Cf += result; mode 2: Cb = gelu(result).
// Block = 256 threads (8 waves) covering 128(M) x 64(N).  Per 64-wide stage:
// wave 0 issues two tensor_load_to_lds (A panel 128x64, B panel 64x64) for
// stage s+1, waits TENSORcnt<=2 (current pair resident, prefetch in flight),
// barrier, then each wave runs 2 sub-steps x 4 v_wmma_f32_16x16x32_bf16 with
// fragments from ds_load_b128 (B panel shared by all 8 waves).
// ---------------------------------------------------------------------------
#define KSTEP 64
#define LDS_A_ELEMS (128 * KSTEP)
#define LDS_B_ELEMS (64 * KSTEP)
#define GEMM_LDS_BYTES ((2 * LDS_A_ELEMS + 2 * LDS_B_ELEMS) * 2)

__global__ void k_wmma_gemm(const bf16_t* __restrict__ A, const bf16_t* __restrict__ W,
                            float* __restrict__ Cf, bf16_t* __restrict__ Cb,
                            int M, int N, int K, int mode) {
    extern __shared__ char smem[];
    bf16_t* ldsA = (bf16_t*)smem;                                  // [2][128*64]
    bf16_t* ldsB = (bf16_t*)(smem + 2 * LDS_A_ELEMS * 2);          // [2][64*64]

    const int lane = threadIdx.x & 31;
    const int wave = threadIdx.x >> 5;
    const int half = lane >> 4;
    const int l15  = lane & 15;
    const int mrow0 = blockIdx.y * 128;
    const int m0    = mrow0 + wave * 16;
    const int n0    = blockIdx.x * 64;

    const v8f vzero = {0.f, 0.f, 0.f, 0.f, 0.f, 0.f, 0.f, 0.f};
    v8f acc[4];
    #pragma unroll
    for (int i = 0; i < 4; ++i) acc[i] = vzero;

    const int kbA = 8 * half;     // A frag: elems 0..7 at k=kbA, 8..15 at k=16+kbA
    const int kbB = 8 * half;     // B frag: elems 0..7 at k=kbB, 8..15 at k=16+kbB
    const int nstage = K / KSTEP;

    if (wave == 0) {
        tdm_load_tile_2d(0u, A + (size_t)mrow0 * K,
                         (unsigned)K, (unsigned)K, (unsigned)M, KSTEP, 128u);
        tdm_load_tile_2d((unsigned)(2 * LDS_A_ELEMS * 2), W + (size_t)n0 * K,
                         (unsigned)K, (unsigned)K, (unsigned)N, KSTEP, 64u);
    }

    for (int s = 0; s < nstage; ++s) {
        const int cur = s & 1;
        if (wave == 0) {
            if (s + 1 < nstage) {
                const int nxt = cur ^ 1;
                const int k1  = (s + 1) * KSTEP;
                tdm_load_tile_2d((unsigned)(nxt * LDS_A_ELEMS * 2),
                                 A + (size_t)mrow0 * K + k1,
                                 (unsigned)K, (unsigned)K, (unsigned)M, KSTEP, 128u);
                tdm_load_tile_2d((unsigned)((2 * LDS_A_ELEMS + nxt * LDS_B_ELEMS) * 2),
                                 W + (size_t)n0 * K + k1,
                                 (unsigned)K, (unsigned)K, (unsigned)N, KSTEP, 64u);
                __builtin_amdgcn_s_wait_tensorcnt((short)2);   // current pair resident
            } else {
                __builtin_amdgcn_s_wait_tensorcnt((short)0);
            }
        }
        __syncthreads();

        const bf16_t* ab = ldsA + cur * LDS_A_ELEMS + (wave * 16 + l15) * KSTEP;
        #pragma unroll
        for (int ks = 0; ks < 2; ++ks) {
            Frag a;
            a.q[0] = *(const uint4*)(ab + ks * 32 + kbA);
            a.q[1] = *(const uint4*)(ab + ks * 32 + 16 + kbA);
            #pragma unroll
            for (int ec = 0; ec < 4; ++ec) {
                Frag b;
                const bf16_t* br = ldsB + cur * LDS_B_ELEMS
                                 + (ec * 16 + l15) * KSTEP + ks * 32 + kbB;
                b.q[0] = *(const uint4*)(br);
                b.q[1] = *(const uint4*)(br + 16);
                acc[ec] = wmma_bf16(a, b, acc[ec]);
            }
        }
        __syncthreads();   // reads done before this buffer is overwritten
    }

    #pragma unroll
    for (int ec = 0; ec < 4; ++ec) {
        #pragma unroll
        for (int r = 0; r < 8; ++r) {
            int m = m0 + r + 8 * half;           // C layout: m = vgpr + 8*(lane>=16)
            int n = n0 + ec * 16 + l15;          //           n = lane & 15
            size_t idx = (size_t)m * N + n;
            float vv = acc[ec][r];
            if (mode == 1) {
                Cf[idx] += vv;
            } else if (mode == 2) {
                float t = vv + 0.044715f * vv * vv * vv;
                float gl = vv * (0.5f * (1.0f + tanhf(0.7978845608f * t)));
                Cb[idx] = (bf16_t)gl;
            } else {
                Cb[idx] = (bf16_t)vv;
            }
        }
    }
}

// ---------------------------------------------------------------------------
// Flash-style causal attention with TDM-staged K/V.
// Grid = (S/64, H, B), block = 128 (4 waves); each wave owns a 16-query tile.
// Key loop is BLOCK-uniform (t0 < qb+64): wave 0 TDM-loads the 32x64 K and V
// tiles (row stride 768) into double-buffered LDS, barrier, all waves
// compute.  Iterations fully beyond a wave's causal range are exact no-ops
// (probabilities underflow to 0).  P is staged through per-wave LDS to build
// the PV A-fragment.
// ---------------------------------------------------------------------------
#define ATT_KV_ELEMS  (32 * 64)                       // one K or V tile
#define ATT_LDS_BYTES ((4 * ATT_KV_ELEMS + 4 * 16 * 32) * 2)   // 2xK + 2xV + 4xP

__global__ void k_attention(const bf16_t* __restrict__ qm, const bf16_t* __restrict__ km,
                            const bf16_t* __restrict__ vm, bf16_t* __restrict__ om) {
    extern __shared__ char smem[];
    bf16_t* ldsK = (bf16_t*)smem;                                    // [2][32*64]
    bf16_t* ldsV = (bf16_t*)(smem + 2 * ATT_KV_ELEMS * 2);           // [2][32*64]
    bf16_t* ldsP = (bf16_t*)(smem + 4 * ATT_KV_ELEMS * 2);           // [4][16*32]

    const int lane = threadIdx.x & 31;
    const int wave = threadIdx.x >> 5;
    const int half = lane >> 4;
    const int l15  = lane & 15;
    const int b  = blockIdx.z;
    const int hh = blockIdx.y;
    const int qb = blockIdx.x * 64;
    const int qs = qb + wave * 16;

    const size_t base = (size_t)b * SEQ * D_MODEL + (size_t)hh * D_HEAD;
    const float scale = 0.125f;  // 1/sqrt(64)

    // Q fragments: 16 queries x 64 e, two 16x32 A fragments (global, once).
    Frag aq[2];
    {
        const bf16_t* qrow = qm + base + (size_t)(qs + l15) * D_MODEL;
        const int kb = 8 * half;
        #pragma unroll
        for (int t = 0; t < 2; ++t) {
            aq[t].q[0] = *(const uint4*)(qrow + t * 32 + kb);
            aq[t].q[1] = *(const uint4*)(qrow + t * 32 + 16 + kb);
        }
    }

    const v8f vzero = {0.f, 0.f, 0.f, 0.f, 0.f, 0.f, 0.f, 0.f};
    v8f oacc[4];
    #pragma unroll
    for (int i = 0; i < 4; ++i) oacc[i] = vzero;
    float mrow[8], lrow[8];
    #pragma unroll
    for (int r = 0; r < 8; ++r) { mrow[r] = -1.0e30f; lrow[r] = 0.f; }

    const int nstage = (qb + 64) >> 5;   // block-uniform #key blocks of 32

    if (wave == 0) {
        tdm_load_tile_2d(0u, km + base, (unsigned)D_HEAD, (unsigned)D_MODEL,
                         (unsigned)SEQ, 64u, 32u);
        tdm_load_tile_2d((unsigned)(2 * ATT_KV_ELEMS * 2), vm + base,
                         (unsigned)D_HEAD, (unsigned)D_MODEL, (unsigned)SEQ, 64u, 32u);
    }

    for (int s = 0; s < nstage; ++s) {
        const int cur = s & 1;
        const int t0  = s * 32;
        if (wave == 0) {
            if (s + 1 < nstage) {
                const int nxt = cur ^ 1;
                const size_t goff = base + (size_t)(t0 + 32) * D_MODEL;
                tdm_load_tile_2d((unsigned)(nxt * ATT_KV_ELEMS * 2), km + goff,
                                 (unsigned)D_HEAD, (unsigned)D_MODEL, (unsigned)SEQ,
                                 64u, 32u);
                tdm_load_tile_2d((unsigned)((2 * ATT_KV_ELEMS + nxt * ATT_KV_ELEMS) * 2),
                                 vm + goff, (unsigned)D_HEAD, (unsigned)D_MODEL,
                                 (unsigned)SEQ, 64u, 32u);
                __builtin_amdgcn_s_wait_tensorcnt((short)2);
            } else {
                __builtin_amdgcn_s_wait_tensorcnt((short)0);
            }
        }
        __syncthreads();

        // ---- scores: two 16x16 tiles (keys t0..+15, t0+16..+31), K=64 ----
        const bf16_t* kt = ldsK + cur * ATT_KV_ELEMS;
        v8f sc0 = vzero, sc1 = vzero;
        #pragma unroll
        for (int t = 0; t < 2; ++t) {
            const int eb = t * 32 + 8 * half;
            Frag bk;
            const bf16_t* kr0 = kt + l15 * 64 + eb;          // key row l15
            bk.q[0] = *(const uint4*)(kr0);
            bk.q[1] = *(const uint4*)(kr0 + 16);
            sc0 = wmma_bf16(aq[t], bk, sc0);
            const bf16_t* kr1 = kt + (16 + l15) * 64 + eb;   // key row 16+l15
            bk.q[0] = *(const uint4*)(kr1);
            bk.q[1] = *(const uint4*)(kr1 + 16);
            sc1 = wmma_bf16(aq[t], bk, sc1);
        }

        // ---- online softmax update ----
        float p0[8], p1[8];
        #pragma unroll
        for (int r = 0; r < 8; ++r) {
            const int qidx = qs + r + 8 * half;
            float s0 = sc0[r] * scale;
            float s1 = sc1[r] * scale;
            if (t0 + l15 > qidx)      s0 = NEGV;
            if (t0 + 16 + l15 > qidx) s1 = NEGV;
            float mx = fmaxf(s0, s1);
            #pragma unroll
            for (int off = 8; off >= 1; off >>= 1) mx = fmaxf(mx, __shfl_xor(mx, off));
            const float mnew  = fmaxf(mrow[r], mx);
            const float alpha = __expf(mrow[r] - mnew);
            const float e0 = __expf(s0 - mnew);
            const float e1 = __expf(s1 - mnew);
            float sum = e0 + e1;
            #pragma unroll
            for (int off = 8; off >= 1; off >>= 1) sum += __shfl_xor(sum, off);
            lrow[r] = lrow[r] * alpha + sum;
            mrow[r] = mnew;
            #pragma unroll
            for (int ec = 0; ec < 4; ++ec) oacc[ec][r] *= alpha;
            p0[r] = e0;
            p1[r] = e1;
        }

        // ---- stage P (16x32) through per-wave LDS -> A fragment ----
        bf16_t* lp = ldsP + wave * (16 * 32);
        #pragma unroll
        for (int r = 0; r < 8; ++r) {
            const int m = r + 8 * half;
            lp[m * 32 + l15]      = (bf16_t)p0[r];
            lp[m * 32 + 16 + l15] = (bf16_t)p1[r];
        }
        __builtin_amdgcn_wave_barrier();  // LDS in-order per wave

        Frag ap;
        {
            const bf16_t* pr = lp + l15 * 32;
            const int kb = 16 * half;
            ap.q[0] = *(const uint4*)(pr + kb);
            ap.q[1] = *(const uint4*)(pr + 8 + kb);
        }

        // ---- O += P(16x32) * V(32x64) from staged LDS tile ----
        const bf16_t* vt = ldsV + cur * ATT_KV_ELEMS;
        #pragma unroll
        for (int ec = 0; ec < 4; ++ec) {
            Frag bv;
            #pragma unroll
            for (int j = 0; j < 16; ++j) {
                bv.h[j] = vt[(16 * half + j) * 64 + ec * 16 + l15];
            }
            oacc[ec] = wmma_bf16(ap, bv, oacc[ec]);
        }
        __syncthreads();   // all reads of this K/V buffer done before overwrite
    }

    // ---- normalize and store ----
    #pragma unroll
    for (int ec = 0; ec < 4; ++ec) {
        #pragma unroll
        for (int r = 0; r < 8; ++r) {
            const int m = r + 8 * half;
            const float vv = oacc[ec][r] / lrow[r];
            om[base + (size_t)(qs + m) * D_MODEL + ec * 16 + l15] = (bf16_t)vv;
        }
    }
}

// ---------------------------------------------------------------------------
// Final-position LayerNorm: one block per batch element.
// ---------------------------------------------------------------------------
__global__ void k_final_ln(const float* __restrict__ x, const int* __restrict__ pidx,
                           const float* __restrict__ g, const float* __restrict__ be,
                           float* __restrict__ normed) {
    __shared__ float red[256];
    const int b   = blockIdx.x;
    const int tid = threadIdx.x;
    const int row = b * SEQ + pidx[0];
    const float* xr = x + (size_t)row * D_MODEL;

    float v[3];
    float s = 0.f;
    #pragma unroll
    for (int i = 0; i < 3; ++i) { v[i] = xr[tid + i * 256]; s += v[i]; }
    red[tid] = s;
    __syncthreads();
    for (int st = 128; st > 0; st >>= 1) {
        if (tid < st) red[tid] += red[tid + st];
        __syncthreads();
    }
    float mean = red[0] * (1.0f / D_MODEL);
    __syncthreads();
    float s2 = 0.f;
    #pragma unroll
    for (int i = 0; i < 3; ++i) { float dd = v[i] - mean; s2 += dd * dd; }
    red[tid] = s2;
    __syncthreads();
    for (int st = 128; st > 0; st >>= 1) {
        if (tid < st) red[tid] += red[tid + st];
        __syncthreads();
    }
    float rstd = rsqrtf(red[0] * (1.0f / D_MODEL) + 1e-5f);
    #pragma unroll
    for (int i = 0; i < 3; ++i) {
        int c = tid + i * 256;
        normed[(size_t)b * D_MODEL + c] = g[c] * ((v[i] - mean) * rstd) + be[c];
    }
}

// ---------------------------------------------------------------------------
// Logits: out[b, v] = normed[b,:] . unembed[v,:].  M=4 only -> plain dot.
// ---------------------------------------------------------------------------
__global__ void k_logits(const float* __restrict__ normed, const float* __restrict__ unembed,
                         float* __restrict__ out) {
    __shared__ float ln[BATCH * D_MODEL];
    for (int i = threadIdx.x; i < BATCH * D_MODEL; i += blockDim.x) ln[i] = normed[i];
    __syncthreads();
    const int v = blockIdx.x * blockDim.x + threadIdx.x;
    if (v >= VOCAB) return;
    const float* ur = unembed + (size_t)v * D_MODEL;
    float a0 = 0.f, a1 = 0.f, a2 = 0.f, a3 = 0.f;
    for (int d = 0; d < D_MODEL; ++d) {
        const float u = ur[d];
        a0 += ln[d] * u;
        a1 += ln[D_MODEL + d] * u;
        a2 += ln[2 * D_MODEL + d] * u;
        a3 += ln[3 * D_MODEL + d] * u;
    }
    out[v]             = a0;
    out[VOCAB + v]     = a1;
    out[2 * VOCAB + v] = a2;
    out[3 * VOCAB + v] = a3;
}

// ---------------------------------------------------------------------------
// Host orchestration
// ---------------------------------------------------------------------------
extern "C" void kernel_launch(void* const* d_in, const int* in_sizes, int n_in,
                              void* d_out, int out_size, void* d_ws, size_t ws_size,
                              hipStream_t stream) {
    (void)in_sizes; (void)n_in; (void)out_size; (void)ws_size;

    const int*   tokens    = (const int*)d_in[0];
    const int*   pidx      = (const int*)d_in[1];
    const float* embedding = (const float*)d_in[2];
    const float* ln1_g     = (const float*)d_in[3];
    const float* ln1_b     = (const float*)d_in[4];
    const float* wq        = (const float*)d_in[5];
    const float* wk        = (const float*)d_in[6];
    const float* wv        = (const float*)d_in[7];
    const float* wo        = (const float*)d_in[8];
    const float* ln2_g     = (const float*)d_in[9];
    const float* ln2_b     = (const float*)d_in[10];
    const float* fc1       = (const float*)d_in[11];
    const float* fc2       = (const float*)d_in[12];
    const float* lnf_g     = (const float*)d_in[13];
    const float* lnf_b     = (const float*)d_in[14];
    const float* unembed   = (const float*)d_in[15];
    float* out = (float*)d_out;

    const int TOK  = BATCH * SEQ;                 // 4096 rows
    const int WSQ  = N_HEADS * D_HEAD * D_MODEL;  // 589824
    const int WFF  = D_FF * D_MODEL;              // 2359296

    char* p = (char*)d_ws;
    auto take = [&](size_t bytes) -> void* {
        void* r = (void*)p;
        p += (bytes + 255) & ~(size_t)255;
        return r;
    };
    float*  x      = (float*)take((size_t)TOK * D_MODEL * sizeof(float));
    bf16_t* h      = (bf16_t*)take((size_t)TOK * D_MODEL * sizeof(bf16_t));
    bf16_t* qb     = (bf16_t*)take((size_t)TOK * D_MODEL * sizeof(bf16_t));
    bf16_t* kb     = (bf16_t*)take((size_t)TOK * D_MODEL * sizeof(bf16_t));
    bf16_t* vb     = (bf16_t*)take((size_t)TOK * D_MODEL * sizeof(bf16_t));
    bf16_t* attnb  = (bf16_t*)take((size_t)TOK * D_MODEL * sizeof(bf16_t));
    bf16_t* ffb    = (bf16_t*)take((size_t)TOK * D_FF * sizeof(bf16_t));
    bf16_t* wqb    = (bf16_t*)take((size_t)WSQ * sizeof(bf16_t));
    bf16_t* wkb    = (bf16_t*)take((size_t)WSQ * sizeof(bf16_t));
    bf16_t* wvb    = (bf16_t*)take((size_t)WSQ * sizeof(bf16_t));
    bf16_t* wob    = (bf16_t*)take((size_t)WSQ * sizeof(bf16_t));
    bf16_t* fc1b   = (bf16_t*)take((size_t)WFF * sizeof(bf16_t));
    bf16_t* fc2b   = (bf16_t*)take((size_t)WFF * sizeof(bf16_t));
    float*  normed = (float*)take((size_t)BATCH * D_MODEL * sizeof(float));

    auto cvt = [&](const float* s, bf16_t* d, int n) {
        k_to_bf16<<<(n + 255) / 256, 256, 0, stream>>>(s, d, n);
    };
    auto gemm = [&](const bf16_t* A, const bf16_t* W, float* Cf, bf16_t* Cb,
                    int M, int N, int K, int mode) {
        dim3 grid(N / 64, M / 128);
        k_wmma_gemm<<<grid, 256, GEMM_LDS_BYTES, stream>>>(A, W, Cf, Cb, M, N, K, mode);
    };

    k_embed<<<(TOK * D_MODEL + 255) / 256, 256, 0, stream>>>(tokens, embedding, x);

    for (int l = 0; l < N_LAYERS; ++l) {
        cvt(wq + (size_t)l * WSQ, wqb, WSQ);
        cvt(wk + (size_t)l * WSQ, wkb, WSQ);
        cvt(wv + (size_t)l * WSQ, wvb, WSQ);
        cvt(wo + (size_t)l * WSQ, wob, WSQ);
        cvt(fc1 + (size_t)l * WFF, fc1b, WFF);
        cvt(fc2 + (size_t)l * WFF, fc2b, WFF);

        k_layernorm<<<TOK, 256, 0, stream>>>(x, ln1_g + l * D_MODEL, ln1_b + l * D_MODEL, h);

        gemm(h, wqb, nullptr, qb, TOK, D_MODEL, D_MODEL, 0);
        gemm(h, wkb, nullptr, kb, TOK, D_MODEL, D_MODEL, 0);
        gemm(h, wvb, nullptr, vb, TOK, D_MODEL, D_MODEL, 0);

        dim3 agrid(SEQ / 64, N_HEADS, BATCH);
        k_attention<<<agrid, 128, ATT_LDS_BYTES, stream>>>(qb, kb, vb, attnb);

        gemm(attnb, wob, x, nullptr, TOK, D_MODEL, D_MODEL, 1);

        k_layernorm<<<TOK, 256, 0, stream>>>(x, ln2_g + l * D_MODEL, ln2_b + l * D_MODEL, h);
        gemm(h, fc1b, nullptr, ffb, TOK, D_FF, D_MODEL, 2);
        gemm(ffb, fc2b, x, nullptr, TOK, D_MODEL, D_FF, 1);
    }

    k_final_ln<<<BATCH, 256, 0, stream>>>(x, pidx, lnf_g, lnf_b, normed);
    k_logits<<<(VOCAB + 255) / 256, 256, 0, stream>>>(normed, unembed, out);
}
